// RecurrentNeuralNetwork_90005334655184
// MI455X (gfx1250) — matrix operand order repaired and checked by reading, not compiled, not run on hardware
//
#include <hip/hip_runtime.h>

// ---------------------------------------------------------------------------
// RNN scan for MI455X (gfx1250, wave32, WMMA).
//   B=256, T=1024, NIN=4, NHID=256, NOUT=2
//   tmp   = x_t @ W_in^T + tanh(h) @ W_hh^T + b
//   h'    = (1-a)h + a*tmp + noise*0.05*sqrt(a)
//   out   = 20*tanh(h' @ W_out^T)
// One workgroup per 16 batch rows (one WMMA M-tile); 16 waves, each owning one
// 16-column N-tile with W_hh held in registers as bf16 B-fragments for the
// whole 1024-step scan. Hidden state lives in LDS in fp32.
// ---------------------------------------------------------------------------

typedef __bf16 bf16_t;
typedef __attribute__((ext_vector_type(16))) __bf16 v16bf;
typedef __attribute__((ext_vector_type(8)))  __bf16 v8bf;
typedef __attribute__((ext_vector_type(8)))  float  v8f;

#define B_SZ   256
#define T_SZ   1024
#define NIN    4
#define NHID   256
#define NOUT   2
#define ROWS   16            // batch rows per workgroup (one WMMA M-tile)
#define NTHREADS 512         // 16 waves
#define ALPHA_V       0.25f
#define NOISE_SCALE_V 0.025f // 0.05 * sqrt(0.25)

__global__ __launch_bounds__(NTHREADS, 1)
void rnn_scan_wmma_kernel(const float* __restrict__ x,        // [B,T,NIN]
                          const float* __restrict__ h0,       // [B,NHID]
                          const float* __restrict__ w_in,     // [NHID,NIN]
                          const float* __restrict__ w_hh,     // [NHID,NHID]
                          const float* __restrict__ bias,     // [NHID]
                          const float* __restrict__ w_out,    // [NOUT,NHID]
                          const float* __restrict__ noise,    // [T,B,NHID]
                          float* __restrict__ hid_list,       // [B,T,NHID]
                          float* __restrict__ out_list,       // [B,T,NOUT]
                          float* __restrict__ hid_final)      // [B,NHID]
{
    __shared__ float                 hs[ROWS * NHID];        // fp32 hidden state
    __shared__ __align__(16) bf16_t  As[ROWS * NHID];        // tanh(h) as bf16
    __shared__ float                 win_s[NHID * NIN];
    __shared__ float                 bias_s[NHID];
    __shared__ float                 wout_s[NOUT * NHID];
    __shared__ float                 xs[ROWS * NIN];

    const int tid  = threadIdx.x;
    const int lane = tid & 31;
    const int wav  = tid >> 5;          // 0..15 : N-tile owner
    const int nlo  = lane & 15;
    const int hi   = lane >> 4;         // half-wave select
    const int row_base = blockIdx.x * ROWS;

    // ---- stage constants + initial hidden state into LDS ----
    for (int i = tid; i < NHID * NIN;  i += NTHREADS) win_s[i]  = w_in[i];
    for (int i = tid; i < NHID;        i += NTHREADS) bias_s[i] = bias[i];
    for (int i = tid; i < NOUT * NHID; i += NTHREADS) wout_s[i] = w_out[i];
    for (int i = tid; i < ROWS * NHID; i += NTHREADS)
        hs[i] = h0[(size_t)(row_base + (i >> 8)) * NHID + (i & 255)];

    // ---- preload this wave's W_hh B-fragments into registers (bf16) ----
    // B 32x16, contiguous k order chosen here (A matches): lane%16 = column n (lanes 0-15: K=0..15, 16-31: K=16..31),
    // halves j=0..15 hold K = hi*16 + j.  tmp[m][n] = sum_k A[m][k]*W_hh[n][k]
    // => B[k][n] = W_hh[n][k]: 16 contiguous f32 along W_hh's row per fragment.
    v16bf bfrag[8];
    {
        const int n = wav * 16 + nlo;
        #pragma unroll
        for (int kt = 0; kt < 8; ++kt) {
            const float* src = &w_hh[(size_t)n * NHID + kt * 32 + hi * 16];
            #pragma unroll
            for (int e = 0; e < 16; ++e)
                bfrag[kt][e] = (bf16_t)src[e];
        }
    }

    __syncthreads();

    for (int t = 0; t < T_SZ; ++t) {
        // 1) shared tanh(h) -> bf16 A tile (each thread converts 8 elems)
        #pragma unroll
        for (int e = 0; e < 8; ++e) {
            const int idx = tid * 8 + e;
            As[idx] = (bf16_t)tanhf(hs[idx]);
        }
        // 2) stage x_t (16 rows x 4)
        if (tid < ROWS * NIN) {
            const int m = tid >> 2, i = tid & 3;
            xs[tid] = x[((size_t)(row_base + m) * T_SZ + t) * NIN + i];
        }
        __syncthreads();

        // 3) WMMA: acc[16x16] = tanh(h_tile) @ W_hh_tile^T  (8 K-steps of 32)
        v8f acc = {};
        #pragma unroll
        for (int kt = 0; kt < 8; ++kt) {
            // A-matrix 16x32 layout: lane m=nlo holds row m; halves 0..15 are
            // K = kt*32 + hi*16 + j (matches B fragment order).
            const v8bf alo = *(const v8bf*)&As[nlo * NHID + kt * 32 + hi * 16];
            const v8bf ahi = *(const v8bf*)&As[nlo * NHID + kt * 32 + hi * 16 + 8];
            v16bf af;
            #pragma unroll
            for (int e = 0; e < 8; ++e) { af[e] = alo[e]; af[e + 8] = ahi[e]; }
            acc = __builtin_amdgcn_wmma_f32_16x16x32_bf16(
                      false, af, false, bfrag[kt], (short)0, acc, false, false);
        }

        // 4) epilogue: + x@W_in^T + bias, leaky update, noise, store hidden_list
        //    C/D layout: VGPR v -> (M = v + 8*hi, N = nlo) per 16x16 f32 table.
        {
            const int n = wav * 16 + nlo;
            const float b_n = bias_s[n];
            #pragma unroll
            for (int v = 0; v < 8; ++v) {
                const int m = v + hi * 8;
                float tmp = acc[v] + b_n;
                #pragma unroll
                for (int i = 0; i < NIN; ++i)
                    tmp += xs[m * NIN + i] * win_s[n * NIN + i];
                const float hold = hs[m * NHID + n];
                const float nz =
                    noise[((size_t)t * B_SZ + row_base + m) * NHID + n];
                const float hnew = (1.0f - ALPHA_V) * hold + ALPHA_V * tmp
                                 + nz * NOISE_SCALE_V;
                hs[m * NHID + n] = hnew;
                hid_list[((size_t)(row_base + m) * T_SZ + t) * NHID + n] = hnew;
            }
            if (t + 1 < T_SZ)   // warm L2 for next step's noise tile
                __builtin_prefetch(
                    &noise[((size_t)(t + 1) * B_SZ + row_base + hi * 8) * NHID + n],
                    0, 0);
        }
        __syncthreads();

        // 5) readout head: out = 20*tanh(h' @ W_out^T)   (32 threads: 16x2 dots)
        if (tid < ROWS * NOUT) {
            const int m = tid >> 1, o = tid & 1;
            float s = 0.0f;
            #pragma unroll 8
            for (int n2 = 0; n2 < NHID; ++n2)
                s += hs[m * NHID + n2] * wout_s[o * NHID + n2];
            out_list[((size_t)(row_base + m) * T_SZ + t) * NOUT + o] =
                20.0f * tanhf(s);
        }
        // next iteration's barrier (after stages 1-2) orders stage-5 hs reads
        // against the next step's hs writes.
    }

    __syncthreads();
    for (int i = tid; i < ROWS * NHID; i += NTHREADS)
        hid_final[(size_t)(row_base + (i >> 8)) * NHID + (i & 255)] = hs[i];
}

extern "C" void kernel_launch(void* const* d_in, const int* in_sizes, int n_in,
                              void* d_out, int out_size, void* d_ws, size_t ws_size,
                              hipStream_t stream) {
    (void)in_sizes; (void)n_in; (void)out_size; (void)d_ws; (void)ws_size;

    const float* x      = (const float*)d_in[0];  // input_signal [B,T,NIN]
    const float* h0     = (const float*)d_in[1];  // hidden       [B,NHID]
    const float* w_in   = (const float*)d_in[2];  // w_in_w       [NHID,NIN]
    const float* w_hh   = (const float*)d_in[3];  // w_hh_w       [NHID,NHID]
    const float* b_hh   = (const float*)d_in[4];  // w_hh_b       [NHID]
    const float* w_out  = (const float*)d_in[5];  // w_out_w      [NOUT,NHID]
    const float* noise  = (const float*)d_in[6];  // noise        [T,B,NHID]

    float* hid_list  = (float*)d_out;                               // [B,T,NHID]
    float* out_list  = hid_list + (size_t)B_SZ * T_SZ * NHID;       // [B,T,NOUT]
    float* hid_final = out_list + (size_t)B_SZ * T_SZ * NOUT;       // [B,NHID]

    dim3 grid(B_SZ / ROWS);      // 16 workgroups
    dim3 block(NTHREADS);        // 16 waves of 32
    rnn_scan_wmma_kernel<<<grid, block, 0, stream>>>(
        x, h0, w_in, w_hh, b_hh, w_out, noise, hid_list, out_list, hid_final);
}
